// MultipleAttentionModel_56281251446913
// MI455X (gfx1250) — matrix operand order repaired and checked by reading, not compiled, not run on hardware
//
#include <hip/hip_runtime.h>
#include <hip/hip_bf16.h>
#include <math.h>

static constexpr int kBS = 64;
static constexpr int kT  = 128;
static constexpr int kG  = 36;
static constexpr int kC  = 256;
static constexpr int kH  = 256;
static constexpr int kQ  = 16;
static constexpr int kNC = 10;

typedef __attribute__((ext_vector_type(16))) _Float16 v16h;
typedef __attribute__((ext_vector_type(8)))  float    v8f;

// Branch-free transcendentals on top of hardware V_EXP_F32 (TRANS32 co-issue).
__device__ __forceinline__ float fast_tanh(float x) {
  float ax = fabsf(x);
  float e  = __expf(-2.0f * ax);          // -> 0 for large |x|
  float t  = (1.0f - e) / (1.0f + e);     // tanh(|x|), saturates to 1
  return copysignf(t, x);
}
__device__ __forceinline__ float sigm(float x) { return 1.0f / (1.0f + __expf(-x)); }

// ---- WMMA fragment helpers (CDNA5 16x16x32 f16, wave32) ---------------------
// A-matrix 16x32 (MxK) 16-bit layout (ISA 7.12.2):
//   lanes 0-15: M=lane,  comps 0-7 -> K=0..7,  comps 8-15 -> K=16..23
//   lanes16-31: M=lane-16, comps 0-7 -> K=8..15, comps 8-15 -> K=24..31
__device__ __forceinline__ v16h load_a_ptr(const float* rowp, int kc, int lane) {
  const float* r = rowp + kc * 32;
  int hi = (lane >> 4) << 3;
  v16h a;
#pragma unroll
  for (int i = 0; i < 8; ++i) a[i] = (_Float16)r[hi + i];
#pragma unroll
  for (int i = 0; i < 8; ++i) a[8 + i] = (_Float16)r[16 + hi + i];
  return a;
}

__device__ __forceinline__ v16h load_a_rm(const float* X, int ldk, int mt, int kc, int lane) {
  return load_a_ptr(X + (size_t)(mt * 16 + (lane & 15)) * ldk, kc, lane);
}

// Packed B fragments: frag f = nt*(K/32)+kc, 512 halves each, lane-major:
//   dst[f*512 + lane*16 + c] = W[nt*16 + (lane&15)][kc*32 + ((lane>=16)?8:0) + (c&7) + ((c>=8)?16:0)]
// so B[k,n] = W[n,k] and the WMMA computes X @ W^T.
__device__ __forceinline__ v8f gemm_acc(const float* A, int ldk, int Kdim, int mt, int nt,
                                        const _Float16* Wp, int lane, v8f acc) {
  const int KC = Kdim >> 5;
  const v16h* bp = (const v16h*)(Wp + (size_t)nt * KC * 512) + lane;
  for (int kc = 0; kc < KC; ++kc) {
    v16h a = load_a_rm(A, ldk, mt, kc, lane);
    v16h b = bp[(size_t)kc * 32];
    acc = __builtin_amdgcn_wmma_f32_16x16x32_f16(false, a, false, b, (short)0, acc,
                                                 false, false);
  }
  return acc;
}

// C/D f32 layout: VGPR r: lanes0-15 M=r, lanes16-31 M=r+8; N = lane&15
__device__ __forceinline__ void store_tile(float* O, int ldo, int mt, int nt, int lane,
                                           v8f acc) {
  int n  = nt * 16 + (lane & 15);
  int mb = mt * 16 + ((lane >> 4) << 3);
#pragma unroll
  for (int r = 0; r < 8; ++r) O[(size_t)(mb + r) * ldo + n] = acc[r];
}

// ---- weight packing: f32 (N x K) row-major -> f16 B fragments ---------------
__global__ void pack_w_kernel(const float* __restrict__ W, _Float16* __restrict__ dst,
                              int N, int K, int srcN) {
  int tid = blockIdx.x * blockDim.x + threadIdx.x;
  int total = N * K;
  if (tid >= total) return;
  int f = tid >> 9;
  int r = tid & 511;
  int lane = r >> 4;
  int c = r & 15;
  int KC = K >> 5;
  int nt = f / KC, kc = f - nt * KC;
  int n = nt * 16 + (lane & 15);
  int k = kc * 32 + ((lane >> 4) << 3) + (c & 7) + ((c >> 3) << 4);
  float v = (n < srcN) ? W[(size_t)n * K + k] : 0.0f;
  dst[tid] = (_Float16)v;
}

// ---- init: mean gaze -> h0, c0 (double-tanh MLPs), Hp for step 0 ------------
__global__ void init_kernel(const float* __restrict__ gaze,
                            const float* __restrict__ ihw1, const float* __restrict__ ihb1,
                            const float* __restrict__ ihw2, const float* __restrict__ ihb2,
                            const float* __restrict__ icw1, const float* __restrict__ icb1,
                            const float* __restrict__ icw2, const float* __restrict__ icb2,
                            const float* __restrict__ sawl, const float* __restrict__ sabl,
                            float* __restrict__ h, float* __restrict__ cst,
                            float* __restrict__ Hp) {
  __shared__ float mg[3];
  __shared__ float a1[kH];
  __shared__ float hrow[kH];
  int b = blockIdx.x, p = threadIdx.x;
  if (p < 3) {
    float s = 0.f;
    for (int tt = 0; tt < kT; ++tt) s += gaze[((size_t)b * kT + tt) * 3 + p];
    mg[p] = s / (float)kT;
  }
  __syncthreads();
  a1[p] = fast_tanh(ihb1[p] + ihw1[p * 3 + 0] * mg[0] + ihw1[p * 3 + 1] * mg[1] +
                    ihw1[p * 3 + 2] * mg[2]);
  __syncthreads();
  {
    float z = ihb2[p];
    const float* wr = ihw2 + (size_t)p * kH;
    for (int k = 0; k < kH; ++k) z += wr[k] * a1[k];
    float hv = fast_tanh(fast_tanh(z));
    hrow[p] = hv;
    h[(size_t)b * kH + p] = hv;
  }
  __syncthreads();
  a1[p] = fast_tanh(icb1[p] + icw1[p * 3 + 0] * mg[0] + icw1[p * 3 + 1] * mg[1] +
                    icw1[p * 3 + 2] * mg[2]);
  __syncthreads();
  {
    float z = icb2[p];
    const float* wr = icw2 + (size_t)p * kH;
    for (int k = 0; k < kH; ++k) z += wr[k] * a1[k];
    cst[(size_t)b * kH + p] = fast_tanh(fast_tanh(z));
  }
  {
    float z = sabl[p];
    const float* wr = sawl + (size_t)p * kH;
    for (int k = 0; k < kH; ++k) z += wr[k] * hrow[k];
    Hp[(size_t)b * kH + p] = z;
  }
}

// ---- K1: spatial attention logits zs, fused Vp GEMM + tanh + wz dot ---------
// grid 144 blocks x 128 threads (4 waves); block owns 16 (b,g) rows.
__global__ void __launch_bounds__(128) spatial_zs_kernel(
    const float* __restrict__ feat, const float* __restrict__ Hp,
    const float* __restrict__ sabc, const float* __restrict__ sawz,
    const _Float16* __restrict__ saWc_p, float* __restrict__ zs, int t) {
  __shared__ float part[4][16];
  const int lane = threadIdx.x & 31;
  const int wid  = threadIdx.x >> 5;
  const int m0   = blockIdx.x * 16;
  const int hi8  = (lane >> 4) << 3;
  const int ln   = lane & 15;

  int mrow = m0 + ln;
  int bA = mrow / kG, gA = mrow - bA * kG;
  const float* arow = feat + ((((size_t)bA * kT + t) * kG + gA) * kC);

  float rs[8];
#pragma unroll
  for (int r = 0; r < 8; ++r) rs[r] = 0.f;

  for (int nt = wid; nt < 16; nt += 4) {
    v8f acc = {};
    const v16h* bp = (const v16h*)(saWc_p + (size_t)nt * 8 * 512) + lane;
    for (int kc = 0; kc < 8; ++kc) {
      v16h a = load_a_ptr(arow, kc, lane);
      v16h bf = bp[(size_t)kc * 32];
      acc = __builtin_amdgcn_wmma_f32_16x16x32_f16(false, a, false, bf, (short)0, acc,
                                                   false, false);
    }
    int n = nt * 16 + ln;
    float wz = sawz[n];
    float bc = sabc[n];
#pragma unroll
    for (int r = 0; r < 8; ++r) {
      int mglob = m0 + r + hi8;
      int bb = mglob / kG;
      float v = fast_tanh(acc[r] + bc + Hp[(size_t)bb * kH + n]) * wz;
      v += __shfl_xor(v, 1, 32);
      v += __shfl_xor(v, 2, 32);
      v += __shfl_xor(v, 4, 32);
      v += __shfl_xor(v, 8, 32);
      rs[r] += v;
    }
  }
  if (ln == 0) {
#pragma unroll
    for (int r = 0; r < 8; ++r) part[wid][r + hi8] = rs[r];
  }
  __syncthreads();
  if (threadIdx.x < 16) {
    zs[m0 + threadIdx.x] = part[0][threadIdx.x] + part[1][threadIdx.x] +
                           part[2][threadIdx.x] + part[3][threadIdx.x];
  }
}

// ---- K2: softmax over grids + weighted feature sum --------------------------
__global__ void spat_kernel(const float* __restrict__ feat, const float* __restrict__ zs,
                            float* __restrict__ spat, int t) {
  __shared__ float sw[kG];
  int b = blockIdx.x;
  if (threadIdx.x == 0) {
    const float* zr = zs + (size_t)b * kG;
    float mx = -1e30f;
    for (int g = 0; g < kG; ++g) mx = fmaxf(mx, zr[g]);
    float s = 0.f;
    for (int g = 0; g < kG; ++g) { float e = __expf(zr[g] - mx); sw[g] = e; s += e; }
    float inv = 1.0f / s;
    for (int g = 0; g < kG; ++g) sw[g] *= inv;
  }
  __syncthreads();
  int cch = threadIdx.x;
  const float* fb = feat + (((size_t)b * kT + t) * kG) * kC + cch;
  float acc = 0.f;
  for (int g = 0; g < kG; ++g) acc += sw[g] * fb[(size_t)g * kC];
  spat[(size_t)b * kC + cch] = acc;
}

// ---- K3: fused recurrent core (queue push, temporal attn, LSTM, MLP, Hp) ----
__global__ void __launch_bounds__(512) core_kernel(
    float* __restrict__ h, float* __restrict__ cst, float* __restrict__ Hp,
    float* __restrict__ spat, float* __restrict__ temp, float* __restrict__ fcb,
    float* __restrict__ y1, float* __restrict__ y2, float* __restrict__ gates,
    float* __restrict__ bufS, float* __restrict__ projQ,
    const float* __restrict__ gaze,
    const float* __restrict__ tabl, const float* __restrict__ tabv,
    const float* __restrict__ tawz,
    const float* __restrict__ bih, const float* __restrict__ bhh,
    const float* __restrict__ wih,
    const float* __restrict__ b1, const float* __restrict__ b2,
    const float* __restrict__ b3, const float* __restrict__ sabl,
    const _Float16* __restrict__ taWl_p, const _Float16* __restrict__ taWv_p,
    const _Float16* __restrict__ whh_p, const _Float16* __restrict__ w1_p,
    const _Float16* __restrict__ w2_p, const _Float16* __restrict__ w3_p,
    const _Float16* __restrict__ saWl_p,
    float* __restrict__ out, int t) {
  __shared__ float ztL[kQ * kBS];
  __shared__ float twL[kQ * kBS];
  const int tid  = threadIdx.x;
  const int lane = tid & 31;
  const int wid  = tid >> 5;            // 16 waves
  const int ln   = lane & 15;
  const int hi8  = (lane >> 4) << 3;
  const int q    = t & 15;

  // Phase a: projQ[q] = h@taWl^T + spat@taWv^T ; bufS[q] = spat
  {
    float* pq = projQ + (size_t)q * kBS * kC;
    for (int tile = wid; tile < 64; tile += 16) {
      int mt = tile >> 4, nt = tile & 15;
      v8f acc = {};
      acc = gemm_acc(h, kC, kC, mt, nt, taWl_p, lane, acc);
      acc = gemm_acc(spat, kC, kC, mt, nt, taWv_p, lane, acc);
      store_tile(pq, kC, mt, nt, lane, acc);
    }
    float* bsd = bufS + (size_t)q * kBS * kC;
    for (int i = tid; i < kBS * kC; i += 512) bsd[i] = spat[i];
  }
  __syncthreads();

  // Phase b: temporal attention over valid queue slots
  int nvalid = (t + 1 < kQ) ? (t + 1) : kQ;
  int s0 = t + 1 - nvalid;
  {
    int pairs = nvalid * kBS;
    for (int pi = tid; pi < pairs; pi += 512) {
      int j = pi >> 6;
      int b = pi & 63;
      int slot = (s0 + j) & 15;
      const float* base = projQ + (size_t)slot * kBS * kC + (size_t)b * kC;
      float z = 0.f;
      for (int p = 0; p < kC; ++p) z += fast_tanh(base[p] + tabl[p] + tabv[p]) * tawz[p];
      ztL[j * kBS + b] = z;
    }
  }
  __syncthreads();
  if (tid < kBS) {
    float mx = -1e30f;
    for (int j = 0; j < nvalid; ++j) mx = fmaxf(mx, ztL[j * kBS + tid]);
    float s = 0.f;
    for (int j = 0; j < nvalid; ++j) {
      float e = __expf(ztL[j * kBS + tid] - mx);
      twL[j * kBS + tid] = e;
      s += e;
    }
    float inv = 1.f / s;
    for (int j = 0; j < nvalid; ++j) twL[j * kBS + tid] *= inv;
  }
  __syncthreads();
  for (int i = tid; i < kBS * kC; i += 512) {
    int b = i >> 8;
    float acc = 0.f;
    for (int j = 0; j < nvalid; ++j) {
      int slot = (s0 + j) & 15;
      acc += twL[j * kBS + b] * bufS[(size_t)slot * kBS * kC + i];
    }
    temp[i] = acc;
  }
  __syncthreads();

  // Phase c: LSTM gates = h@Whh^T + bih + bhh + gaze@Wih^T, then h/c update
  for (int tile = wid; tile < 256; tile += 16) {
    int mt = tile >> 6, nt = tile & 63;
    v8f acc = {};
    acc = gemm_acc(h, kC, kC, mt, nt, whh_p, lane, acc);
    int n = nt * 16 + ln;
    float bb = bih[n] + bhh[n];
    const float* wr = wih + (size_t)n * 3;
#pragma unroll
    for (int r = 0; r < 8; ++r) {
      int b = mt * 16 + r + hi8;
      const float* gz = gaze + ((size_t)b * kT + t) * 3;
      gates[(size_t)b * 1024 + n] =
          acc[r] + bb + wr[0] * gz[0] + wr[1] * gz[1] + wr[2] * gz[2];
    }
  }
  __syncthreads();
  for (int i = tid; i < kBS * kH; i += 512) {
    int b = i >> 8, u = i & 255;
    const float* gr = gates + (size_t)b * 1024;
    float gi_ = gr[u], gf_ = gr[256 + u], gg_ = gr[512 + u], go_ = gr[768 + u];
    float cn = sigm(gf_) * cst[i] + sigm(gi_) * fast_tanh(gg_);
    cst[i] = cn;
    h[i] = sigm(go_) * fast_tanh(cn);
  }
  __syncthreads();

  // Phase d: fc = [temp, h]
  for (int i = tid; i < kBS * 512; i += 512) {
    int b = i >> 9, cc2 = i & 511;
    fcb[i] = (cc2 < 256) ? temp[b * 256 + cc2] : h[b * 256 + (cc2 - 256)];
  }
  __syncthreads();

  // Phase e: MLP classifier
  for (int tile = wid; tile < 128; tile += 16) {
    int mt = tile >> 5, nt = tile & 31;
    v8f acc = {};
    acc = gemm_acc(fcb, 512, 512, mt, nt, w1_p, lane, acc);
    int n = nt * 16 + ln;
    float bb = b1[n];
#pragma unroll
    for (int r = 0; r < 8; ++r) y1[(size_t)(mt * 16 + r + hi8) * 512 + n] = acc[r] + bb;
  }
  __syncthreads();
  for (int tile = wid; tile < 128; tile += 16) {
    int mt = tile >> 5, nt = tile & 31;
    v8f acc = {};
    acc = gemm_acc(y1, 512, 512, mt, nt, w2_p, lane, acc);
    int n = nt * 16 + ln;
    float bb = b2[n];
#pragma unroll
    for (int r = 0; r < 8; ++r) y2[(size_t)(mt * 16 + r + hi8) * 512 + n] = acc[r] + bb;
  }
  __syncthreads();
  if (wid < 4) {
    int mt = wid;
    v8f acc = {};
    acc = gemm_acc(y2, 512, 512, mt, 0, w3_p, lane, acc);
    if (ln < kNC) {
      float bb = b3[ln];
#pragma unroll
      for (int r = 0; r < 8; ++r) {
        int b = mt * 16 + r + hi8;
        out[((size_t)t * kBS + b) * kNC + ln] = acc[r] + bb;
      }
    }
  }
  __syncthreads();

  // Phase f: Hp for next step = h@saWl^T + sabl
  for (int tile = wid; tile < 64; tile += 16) {
    int mt = tile >> 4, nt = tile & 15;
    v8f acc = {};
    acc = gemm_acc(h, kC, kC, mt, nt, saWl_p, lane, acc);
    int n = nt * 16 + ln;
    float bb = sabl[n];
#pragma unroll
    for (int r = 0; r < 8; ++r) Hp[(size_t)(mt * 16 + r + hi8) * kC + n] = acc[r] + bb;
  }
}

extern "C" void kernel_launch(void* const* d_in, const int* in_sizes, int n_in,
                              void* d_out, int out_size, void* d_ws, size_t ws_size,
                              hipStream_t stream) {
  const float* feat = (const float*)d_in[0];
  const float* gaze = (const float*)d_in[1];
  const float* ihw1 = (const float*)d_in[2];
  const float* ihb1 = (const float*)d_in[3];
  const float* ihw2 = (const float*)d_in[4];
  const float* ihb2 = (const float*)d_in[5];
  const float* icw1 = (const float*)d_in[6];
  const float* icb1 = (const float*)d_in[7];
  const float* icw2 = (const float*)d_in[8];
  const float* icb2 = (const float*)d_in[9];
  const float* wih  = (const float*)d_in[10];
  const float* whh  = (const float*)d_in[11];
  const float* bih  = (const float*)d_in[12];
  const float* bhh  = (const float*)d_in[13];
  const float* sawl = (const float*)d_in[14];
  const float* sabl = (const float*)d_in[15];
  const float* sawc = (const float*)d_in[16];
  const float* sabc = (const float*)d_in[17];
  const float* sawz = (const float*)d_in[18];
  const float* tawl = (const float*)d_in[19];
  const float* tabl = (const float*)d_in[20];
  const float* tawv = (const float*)d_in[21];
  const float* tabv = (const float*)d_in[22];
  const float* tawz = (const float*)d_in[23];
  const float* w1   = (const float*)d_in[24];
  const float* b1   = (const float*)d_in[25];
  const float* w2   = (const float*)d_in[26];
  const float* b2   = (const float*)d_in[27];
  const float* w3   = (const float*)d_in[28];
  const float* b3   = (const float*)d_in[29];
  float* out = (float*)d_out;

  // workspace layout (floats, then f16 packed weights) ~5.2 MB total
  float* F = (float*)d_ws;
  float* h     = F + 0;        // 64x256
  float* cst   = F + 16384;    // 64x256
  float* Hp    = F + 32768;    // 64x256
  float* spat  = F + 49152;    // 64x256
  float* temp  = F + 65536;    // 64x256
  float* zs    = F + 81920;    // 64x36
  float* fcb   = F + 84224;    // 64x512
  float* y1    = F + 116992;   // 64x512
  float* y2    = F + 149760;   // 64x512
  float* gates = F + 182528;   // 64x1024
  float* bufS  = F + 248064;   // 16x64x256
  float* projQ = F + 510208;   // 16x64x256
  _Float16* HB = (_Float16*)(F + 772352);
  _Float16* saWl_p = HB + 0;
  _Float16* saWc_p = HB + 65536;
  _Float16* taWl_p = HB + 131072;
  _Float16* taWv_p = HB + 196608;
  _Float16* whh_p  = HB + 262144;
  _Float16* w1_p   = HB + 524288;
  _Float16* w2_p   = HB + 786432;
  _Float16* w3_p   = HB + 1048576;

  pack_w_kernel<<<256, 256, 0, stream>>>(sawl, saWl_p, 256, 256, 256);
  pack_w_kernel<<<256, 256, 0, stream>>>(sawc, saWc_p, 256, 256, 256);
  pack_w_kernel<<<256, 256, 0, stream>>>(tawl, taWl_p, 256, 256, 256);
  pack_w_kernel<<<256, 256, 0, stream>>>(tawv, taWv_p, 256, 256, 256);
  pack_w_kernel<<<1024, 256, 0, stream>>>(whh, whh_p, 1024, 256, 1024);
  pack_w_kernel<<<1024, 256, 0, stream>>>(w1, w1_p, 512, 512, 512);
  pack_w_kernel<<<1024, 256, 0, stream>>>(w2, w2_p, 512, 512, 512);
  pack_w_kernel<<<32, 256, 0, stream>>>(w3, w3_p, 16, 512, 10);

  init_kernel<<<64, 256, 0, stream>>>(gaze, ihw1, ihb1, ihw2, ihb2, icw1, icb1, icw2,
                                      icb2, sawl, sabl, h, cst, Hp);

  for (int t = 0; t < kT; ++t) {
    spatial_zs_kernel<<<144, 128, 0, stream>>>(feat, Hp, sabc, sawz, saWc_p, zs, t);
    spat_kernel<<<64, 256, 0, stream>>>(feat, zs, spat, t);
    core_kernel<<<1, 512, 0, stream>>>(h, cst, Hp, spat, temp, fcb, y1, y2, gates, bufS,
                                       projQ, gaze, tabl, tabv, tawz, bih, bhh, wih, b1,
                                       b2, b3, sabl, taWl_p, taWv_p, whh_p, w1_p, w2_p,
                                       w3_p, saWl_p, out, t);
  }
  (void)in_sizes; (void)n_in; (void)out_size; (void)ws_size;
}
